// Decoder_90847148245046
// MI455X (gfx1250) — matrix operand order repaired and checked by reading, not compiled, not run on hardware
//
#include <hip/hip_runtime.h>

typedef __attribute__((ext_vector_type(16))) __bf16       v16bf;
typedef __attribute__((ext_vector_type(8)))  float        v8f;
typedef __attribute__((ext_vector_type(8)))  unsigned int v8u;

#define B_  64
#define ST_ 1024
#define TT_ 256
#define H_  512
#define E_  512
#define O_  31
#define G3_ 1536

// ---------- helpers ----------
__device__ inline unsigned short f2bf(float x) {
  unsigned u = __builtin_bit_cast(unsigned, x);
  unsigned r = u + 0x7FFFu + ((u >> 16) & 1u);   // RNE
  return (unsigned short)(r >> 16);
}
__device__ inline float bflo(unsigned u) { return __builtin_bit_cast(float, u << 16); }
__device__ inline float bfhi(unsigned u) { return __builtin_bit_cast(float, u & 0xFFFF0000u); }

// A tile 16x32 (MxK), row-major source, ISA 16-bit A layout:
// lane: M = lane&15 ; dword q*4+i -> K = q*16 + (lane>=16)*8 + 2i
__device__ inline v16bf load_a(const unsigned short* base, int ld, int row0, int k0, int lane) {
  v8u r;
  int rl = lane & 15, hi = (lane >> 4) & 1;
  const unsigned short* p = base + (size_t)(row0 + rl) * ld + k0;
#pragma unroll
  for (int q = 0; q < 2; ++q)
#pragma unroll
    for (int i = 0; i < 4; ++i)
      r[q * 4 + i] = *(const unsigned int*)(p + q * 16 + hi * 8 + i * 2);
  return __builtin_bit_cast(v16bf, r);
}

// B tile 32x16 (KxN) taken as rows of W (NxK row-major):
// lane: N = lane&15 ; dword q*4+i -> K = q*16 + (lane>=16)*8 + 2i
__device__ inline v16bf load_b(const unsigned short* base, int ld, int n0, int k0, int lane) {
  v8u r;
  int nl = lane & 15, hi = (lane >> 4) & 1;
  const unsigned short* p = base + (size_t)(n0 + nl) * ld + k0;
#pragma unroll
  for (int i = 0; i < 8; ++i)
    r[i] = *(const unsigned int*)(p + (i >> 2) * 16 + hi * 8 + (i & 3) * 2);
  return __builtin_bit_cast(v16bf, r);
}

__device__ inline v8f wmma_bf16(v16bf a, v16bf b, v8f c) {
  return __builtin_amdgcn_wmma_f32_16x16x32_bf16(false, a, false, b, (short)0, c, false, false);
}

// ---------- prep kernels ----------
__global__ void cvt_bf16(const float* __restrict__ s, unsigned short* __restrict__ d, long long n) {
  long long i = (long long)blockIdx.x * blockDim.x + threadIdx.x;
  long long st = (long long)gridDim.x * blockDim.x;
  for (; i < n; i += st) d[i] = f2bf(s[i]);
}

__global__ void gather_emb(const int* __restrict__ trg, const float* __restrict__ embed,
                           unsigned short* __restrict__ emb_bf) {
  int row = blockIdx.x;           // row = t*64 + b  (time-major)
  int t = row >> 6, b = row & 63;
  int tok = trg[b * TT_ + t];
  const float* src = embed + (size_t)tok * E_;
  unsigned short* dst = emb_bf + (size_t)row * E_;
  for (int e = threadIdx.x; e < E_; e += blockDim.x) dst[e] = f2bf(src[e]);
}

__global__ void init_h(const float* __restrict__ h0, float* __restrict__ hf,
                       unsigned short* __restrict__ hb) {
  int b = blockIdx.x;
  for (int e = threadIdx.x; e < H_; e += blockDim.x) {
    float v = h0[(size_t)b * H_ + e];
    hf[(size_t)b * H_ + e] = v;
    hb[(size_t)b * H_ + e] = f2bf(v);
  }
}

// ---------- GI = EMB @ W_ih^T + b_ih  (16384 x 1536, K=512), WMMA bf16 ----------
__global__ void __launch_bounds__(256) gi_gemm(const unsigned short* __restrict__ A,
                                               const unsigned short* __restrict__ Wb,
                                               const float* __restrict__ bias,
                                               float* __restrict__ GI) {
  int tid = threadIdx.x, lane = tid & 31, w = tid >> 5;
  int mi = w & 3, ni = w >> 2;                   // 4x2 waves -> 64x64 tile
  int row0 = blockIdx.y * 64 + mi * 16;
  int col0 = blockIdx.x * 64 + ni * 32;
  v8f acc0 = {}, acc1 = {};
  for (int k0 = 0; k0 < E_; k0 += 32) {
    v16bf a  = load_a(A,  E_, row0, k0, lane);
    v16bf b0 = load_b(Wb, E_, col0,      k0, lane);
    v16bf b1 = load_b(Wb, E_, col0 + 16, k0, lane);
    acc0 = wmma_bf16(a, b0, acc0);
    acc1 = wmma_bf16(a, b1, acc1);
  }
  int hi8 = (lane >> 4) * 8, nlo = lane & 15;
#pragma unroll
  for (int r = 0; r < 8; ++r) {
    int m = row0 + hi8 + r;
    int c = col0 + nlo;
    GI[(size_t)m * G3_ + c]      = acc0[r] + bias[c];
    GI[(size_t)m * G3_ + c + 16] = acc1[r] + bias[c + 16];
  }
}

// ---------- per-step: gh = h @ W_hh^T, gates, h_new ----------
// grid = 8 blocks, block owns hidden units [j0, j0+64) across all 3 gates.
__global__ void __launch_bounds__(256) gru_step(const unsigned short* __restrict__ hbf_in,
                                                const float* __restrict__ h_in,
                                                const unsigned short* __restrict__ Whh,
                                                const float* __restrict__ b_hh,
                                                const float* __restrict__ gi,      // + t*64*1536
                                                float* __restrict__ h_out,
                                                unsigned short* __restrict__ hbf_out) {
  __shared__ unsigned short shH[B_ * H_];        // 64 KB bf16 A-operand
  int tid = threadIdx.x, lane = tid & 31, w = tid >> 5;

  // async-copy 64KB global -> LDS (Tensor/async path, tracked by ASYNCcnt)
  unsigned lbase = (unsigned)(unsigned long long)(&shH[0]);
  unsigned long long gbase = (unsigned long long)hbf_in;
#pragma unroll
  for (int k = 0; k < 16; ++k) {
    unsigned off = (unsigned)(tid + k * 256) * 16u;
    unsigned la = lbase + off;
    unsigned long long ga = gbase + off;
    asm volatile("global_load_async_to_lds_b128 %0, %1, off" :: "v"(la), "v"(ga) : "memory");
  }
  asm volatile("s_wait_asynccnt 0x0" ::: "memory");
  __syncthreads();

  int mi = w & 3, nh = w >> 2;
  int j0 = blockIdx.x * 64 + nh * 32;
  int row0 = mi * 16;
  v8f acc[3][2];
#pragma unroll
  for (int g = 0; g < 3; ++g) { acc[g][0] = {}; acc[g][1] = {}; }

  for (int k0 = 0; k0 < H_; k0 += 32) {
    v16bf a = load_a(shH, H_, row0, k0, lane);   // ds loads
#pragma unroll
    for (int g = 0; g < 3; ++g) {
      v16bf b0 = load_b(Whh, H_, g * H_ + j0,      k0, lane);
      v16bf b1 = load_b(Whh, H_, g * H_ + j0 + 16, k0, lane);
      acc[g][0] = wmma_bf16(a, b0, acc[g][0]);
      acc[g][1] = wmma_bf16(a, b1, acc[g][1]);
    }
  }

  int hi8 = (lane >> 4) * 8, nlo = lane & 15;
#pragma unroll
  for (int r = 0; r < 8; ++r) {
    int m = row0 + hi8 + r;
    const float* girow = gi + (size_t)m * G3_;
#pragma unroll
    for (int u = 0; u < 2; ++u) {
      int j = j0 + u * 16 + nlo;
      float ghr = acc[0][u][r] + b_hh[j];
      float ghz = acc[1][u][r] + b_hh[H_ + j];
      float ghn = acc[2][u][r] + b_hh[2 * H_ + j];
      float rg = 1.0f / (1.0f + __expf(-(girow[j] + ghr)));
      float zg = 1.0f / (1.0f + __expf(-(girow[H_ + j] + ghz)));
      float ng = tanhf(girow[2 * H_ + j] + rg * ghn);
      float hn = (1.0f - zg) * ng + zg * h_in[(size_t)m * H_ + j];
      h_out[(size_t)m * H_ + j] = hn;
      hbf_out[(size_t)m * H_ + j] = f2bf(hn);
    }
  }
}

// ---------- per-step fused attention + FC, one workgroup per batch row ----------
__device__ inline void acc8(float* a, uint4 v, float ws) {
  a[0] += ws * bflo(v.x); a[1] += ws * bfhi(v.x);
  a[2] += ws * bflo(v.y); a[3] += ws * bfhi(v.y);
  a[4] += ws * bflo(v.z); a[5] += ws * bfhi(v.z);
  a[6] += ws * bflo(v.w); a[7] += ws * bfhi(v.w);
}

__global__ void __launch_bounds__(256) attn_fc(const unsigned short* __restrict__ encbf,
                                               const float* __restrict__ hnew,
                                               const int* __restrict__ source_len,
                                               const int* __restrict__ trg_len,
                                               const float* __restrict__ fcW,
                                               const float* __restrict__ fcb,
                                               float* __restrict__ out, int t) {
  __shared__ float sh_h[H_];
  __shared__ float sh_w[ST_];
  __shared__ float part[8][H_];
  __shared__ float sh_ctx[H_];
  __shared__ float red[256];
  int b = blockIdx.x, tid = threadIdx.x, lane = tid & 31, w = tid >> 5;

  for (int i = tid; i < H_; i += 256) sh_h[i] = hnew[(size_t)b * H_ + i];
  __syncthreads();

  int slen = source_len[b];
  const unsigned short* enc = encbf + (size_t)b * ST_ * H_;

  // scores (masked)
  float sc[4];
#pragma unroll
  for (int i = 0; i < 4; ++i) {
    int s = tid + i * 256;
    const uint4* row = (const uint4*)(enc + (size_t)s * H_);
    float dot = 0.0f;
    for (int c = 0; c < 64; ++c) {
      uint4 v = row[c];
      int h0 = c * 8;
      dot += bflo(v.x) * sh_h[h0]     + bfhi(v.x) * sh_h[h0 + 1]
           + bflo(v.y) * sh_h[h0 + 2] + bfhi(v.y) * sh_h[h0 + 3]
           + bflo(v.z) * sh_h[h0 + 4] + bfhi(v.z) * sh_h[h0 + 5]
           + bflo(v.w) * sh_h[h0 + 6] + bfhi(v.w) * sh_h[h0 + 7];
    }
    sc[i] = (s < slen) ? dot : -1e9f;
  }

  // softmax: max
  float mx = fmaxf(fmaxf(sc[0], sc[1]), fmaxf(sc[2], sc[3]));
  red[tid] = mx; __syncthreads();
  for (int st = 128; st > 0; st >>= 1) {
    if (tid < st) red[tid] = fmaxf(red[tid], red[tid + st]);
    __syncthreads();
  }
  mx = red[0]; __syncthreads();
  // softmax: sum
  float esum = 0.0f;
#pragma unroll
  for (int i = 0; i < 4; ++i) {
    float e = __expf(sc[i] - mx);
    sh_w[tid + i * 256] = e;
    esum += e;
  }
  red[tid] = esum; __syncthreads();
  for (int st = 128; st > 0; st >>= 1) {
    if (tid < st) red[tid] += red[tid + st];
    __syncthreads();
  }
  float inv = 1.0f / red[0]; __syncthreads();
#pragma unroll
  for (int i = 0; i < 4; ++i) sh_w[tid + i * 256] *= inv;
  __syncthreads();

  // ctx: wave w handles s in [w*128, w*128+128); lane owns 16 contiguous h
  float acc[16];
#pragma unroll
  for (int i = 0; i < 16; ++i) acc[i] = 0.0f;
  int hbase = lane * 16;
  for (int s = w * 128; s < w * 128 + 128; ++s) {
    float ws = sh_w[s];
    const uint4* p = (const uint4*)(enc + (size_t)s * H_ + hbase);
    acc8(acc,     p[0], ws);
    acc8(acc + 8, p[1], ws);
  }
#pragma unroll
  for (int i = 0; i < 16; ++i) part[w][hbase + i] = acc[i];
  __syncthreads();
  for (int h = tid; h < H_; h += 256) {
    float s = 0.0f;
#pragma unroll
    for (int ww = 0; ww < 8; ++ww) s += part[ww][h];
    sh_ctx[h] = s;
  }
  __syncthreads();

  // fc over [h_new, ctx], masked output (PAD = 0)
  if (tid < O_) {
    float v = fcb[tid];
    const float* wr = fcW + (size_t)tid * (2 * H_);
    for (int k = 0; k < H_; ++k) v += wr[k] * sh_h[k];
    for (int k = 0; k < H_; ++k) v += wr[H_ + k] * sh_ctx[k];
    out[((size_t)b * TT_ + t) * O_ + tid] = (t < trg_len[b]) ? v : 0.0f;
  }
}

// ---------- host ----------
extern "C" void kernel_launch(void* const* d_in, const int* in_sizes, int n_in,
                              void* d_out, int out_size, void* d_ws, size_t ws_size,
                              hipStream_t stream) {
  const int*   trg_inputs = (const int*)d_in[0];
  const int*   trg_len    = (const int*)d_in[1];
  const int*   source_len = (const int*)d_in[2];
  const float* enc        = (const float*)d_in[3];
  const float* h0         = (const float*)d_in[4];
  const float* embed      = (const float*)d_in[5];
  const float* W_ih       = (const float*)d_in[6];
  const float* W_hh       = (const float*)d_in[7];
  const float* b_ih       = (const float*)d_in[8];
  const float* b_hh       = (const float*)d_in[9];
  const float* fc_W       = (const float*)d_in[10];
  const float* fc_b       = (const float*)d_in[11];
  float* out = (float*)d_out;
  (void)in_sizes; (void)n_in; (void)out_size; (void)ws_size;

  char* ws = (char*)d_ws;
  size_t off = 0;
  auto alloc = [&](size_t bytes) {
    char* p = ws + off;
    off += (bytes + 255) & ~(size_t)255;
    return p;
  };
  unsigned short* encbf = (unsigned short*)alloc((size_t)B_ * ST_ * H_ * 2);
  unsigned short* wihbf = (unsigned short*)alloc((size_t)G3_ * E_ * 2);
  unsigned short* whhbf = (unsigned short*)alloc((size_t)G3_ * H_ * 2);
  unsigned short* embbf = (unsigned short*)alloc((size_t)TT_ * B_ * E_ * 2);
  float*          gi    = (float*)alloc((size_t)TT_ * B_ * G3_ * 4);
  float*          hA    = (float*)alloc((size_t)B_ * H_ * 4);
  float*          hB    = (float*)alloc((size_t)B_ * H_ * 4);
  unsigned short* hAbf  = (unsigned short*)alloc((size_t)B_ * H_ * 2);
  unsigned short* hBbf  = (unsigned short*)alloc((size_t)B_ * H_ * 2);

  cvt_bf16<<<512, 256, 0, stream>>>(enc, encbf, (long long)B_ * ST_ * H_);
  cvt_bf16<<<64, 256, 0, stream>>>(W_ih, wihbf, (long long)G3_ * E_);
  cvt_bf16<<<64, 256, 0, stream>>>(W_hh, whhbf, (long long)G3_ * H_);
  gather_emb<<<TT_ * B_, 256, 0, stream>>>(trg_inputs, embed, embbf);
  init_h<<<B_, 256, 0, stream>>>(h0, hA, hAbf);
  gi_gemm<<<dim3(G3_ / 64, (TT_ * B_) / 64), 256, 0, stream>>>(embbf, wihbf, b_ih, gi);

  for (int t = 0; t < TT_; ++t) {
    const float*          hin    = (t & 1) ? hB : hA;
    const unsigned short* hinbf  = (t & 1) ? hBbf : hAbf;
    float*                hout   = (t & 1) ? hA : hB;
    unsigned short*       houtbf = (t & 1) ? hAbf : hBbf;
    gru_step<<<H_ / 64, 256, 0, stream>>>(hinbf, hin, whhbf, b_hh,
                                          gi + (size_t)t * B_ * G3_, hout, houtbf);
    attn_fc<<<B_, 256, 0, stream>>>(encbf, hout, source_len, trg_len, fc_W, fc_b, out, t);
  }
}
